// TransformConvCrossAttention_7370163880438
// MI455X (gfx1250) — matrix operand order repaired and checked by reading, not compiled, not run on hardware
//
#include <hip/hip_runtime.h>
#include <hip/hip_bf16.h>

typedef __attribute__((ext_vector_type(16))) __bf16 v16bf;
typedef __attribute__((ext_vector_type(8)))  __bf16 v8bf;
typedef __attribute__((ext_vector_type(8)))  float  v8f;
typedef __attribute__((ext_vector_type(4)))  float  f32x4;
typedef __attribute__((ext_vector_type(8)))  unsigned short u16x8;
typedef __attribute__((ext_vector_type(4)))  unsigned short u16x4;
typedef __attribute__((vector_size(16)))     int    i32x4;

#define LDS_AS __attribute__((address_space(3)))

#if defined(__has_builtin)
#if __has_builtin(__builtin_amdgcn_global_load_async_to_lds_b128)
#define HAVE_ASYNC_LDS 1
#endif
#endif
#ifndef HAVE_ASYNC_LDS
#define HAVE_ASYNC_LDS 0
#endif

__device__ __forceinline__ unsigned short f2bf(float x) {
  unsigned int u = __float_as_uint(x);
  u += 0x7FFFu + ((u >> 16) & 1u);          // round-to-nearest-even
  return (unsigned short)(u >> 16);
}
__device__ __forceinline__ float bf2f(unsigned short h) {
  return __uint_as_float(((unsigned int)h) << 16);
}

// 16-byte global -> LDS copy: async DMA (ASYNCcnt) when available.
__device__ __forceinline__ void copy16_g2l(const unsigned short* g, unsigned short* l) {
#if HAVE_ASYNC_LDS
  __builtin_amdgcn_global_load_async_to_lds_b128(
      (i32x4*)g, (LDS_AS i32x4*)l, 0, 0);
#else
  *reinterpret_cast<u16x8*>(l) = *reinterpret_cast<const u16x8*>(g);
#endif
}
__device__ __forceinline__ void wait_async0() {
#if HAVE_ASYNC_LDS
#if __has_builtin(__builtin_amdgcn_s_wait_asynccnt)
  __builtin_amdgcn_s_wait_asynccnt(0);
#else
  asm volatile("s_wait_asynccnt 0x0" ::: "memory");
#endif
#endif
}

// ---------------------------------------------------------------------------
// bf16 WMMA GEMM:  C[M,N] = epi( A[M,K](lda) @ B[N,K](ldb)^T )
// 256x128 block tile, Kc=32, double-buffered LDS fed by async global->LDS DMA.
// 256 threads = 8 waves; each wave computes 64x64 (4x4 16x16x32 WMMAs/step).
// ---------------------------------------------------------------------------
template<typename OT>
__global__ __launch_bounds__(256) void wmma_gemm_bf16(
    const unsigned short* __restrict__ Ag, long aBatch, int lda,
    const unsigned short* __restrict__ Bg, long bBatch, int ldb,
    OT* __restrict__ Cg, long cBatch, int ldc,
    int N, int Kd,
    const float* __restrict__ bias,
    const unsigned char* __restrict__ mask,
    float scale, float negFill)
{
  __shared__ unsigned short Asm[2][256 * 32];   // 16KB per buffer
  __shared__ unsigned short Bsm[2][128 * 32];   //  8KB per buffer

  const int m0 = blockIdx.y * 256;
  const int n0 = blockIdx.x * 128;
  Ag += (size_t)blockIdx.z * (size_t)aBatch;
  Bg += (size_t)blockIdx.z * (size_t)bBatch;
  Cg += (size_t)blockIdx.z * (size_t)cBatch;

  const int t     = threadIdx.x;
  const int lane  = t & 31;
  const int wid   = t >> 5;
  const int wm    = wid & 3;     // m offset = 64*wm
  const int wn    = wid >> 2;    // n offset = 64*wn
  const int lrow  = lane & 15;
  const int lhalf = lane >> 4;

  // Stage one 256x32 A tile and 128x32 B tile into LDS buffer `buf`.
  auto stage = [&](int buf, int kk) {
#pragma unroll
    for (int it = 0; it < 4; ++it) {             // A: 1024 chunks
      int chunk = it * 256 + t;
      int r = chunk >> 2;                        // 0..255
      int c = (chunk & 3) * 8;                   // 0,8,16,24
      copy16_g2l(Ag + (size_t)(m0 + r) * lda + kk + c, &Asm[buf][r * 32 + c]);
    }
#pragma unroll
    for (int it = 0; it < 2; ++it) {             // B: 512 chunks
      int chunk = it * 256 + t;
      int r = chunk >> 2;                        // 0..127
      int c = (chunk & 3) * 8;
      copy16_g2l(Bg + (size_t)(n0 + r) * ldb + kk + c, &Bsm[buf][r * 32 + c]);
    }
  };

  v8f acc[4][4];
#pragma unroll
  for (int mi = 0; mi < 4; ++mi)
#pragma unroll
    for (int ni = 0; ni < 4; ++ni)
#pragma unroll
      for (int e = 0; e < 8; ++e) acc[mi][ni][e] = 0.0f;

  stage(0, 0);
  int ib = 0;
  for (int kk = 0; kk < Kd; kk += 32, ib ^= 1) {
    wait_async0();          // my DMA for buf `ib` complete
    __syncthreads();        // everyone's complete; prior compute done
    if (kk + 32 < Kd) stage(ib ^ 1, kk + 32);   // overlap next tile's DMA

    // ---- fragments from LDS (ISA 7.12.2 layouts) ----
    v16bf afrag[4];
#pragma unroll
    for (int mi = 0; mi < 4; ++mi) {
      int row = wm * 64 + mi * 16 + lrow;
      int kb = lhalf * 8;
      v8bf lo = *reinterpret_cast<const v8bf*>(&Asm[ib][row * 32 + kb]);
      v8bf hi = *reinterpret_cast<const v8bf*>(&Asm[ib][row * 32 + kb + 16]);
      afrag[mi] = __builtin_shufflevector(lo, hi, 0,1,2,3,4,5,6,7,8,9,10,11,12,13,14,15);
    }
    v16bf bfrag[4];
#pragma unroll
    for (int ni = 0; ni < 4; ++ni) {
      int col = wn * 64 + ni * 16 + lrow;
      v8bf blo = *reinterpret_cast<const v8bf*>(&Bsm[ib][col * 32 + lhalf * 8]);
      v8bf bhi = *reinterpret_cast<const v8bf*>(&Bsm[ib][col * 32 + lhalf * 8 + 16]);
      bfrag[ni] = __builtin_shufflevector(blo, bhi, 0,1,2,3,4,5,6,7,8,9,10,11,12,13,14,15);
    }
#pragma unroll
    for (int mi = 0; mi < 4; ++mi)
#pragma unroll
      for (int ni = 0; ni < 4; ++ni)
        acc[mi][ni] = __builtin_amdgcn_wmma_f32_16x16x32_bf16(
            false, afrag[mi], false, bfrag[ni], (short)0, acc[mi][ni], false, false);
  }

  // ---- epilogue ----
#pragma unroll
  for (int mi = 0; mi < 4; ++mi)
#pragma unroll
    for (int ni = 0; ni < 4; ++ni)
#pragma unroll
      for (int r = 0; r < 8; ++r) {
        int grow = m0 + wm * 64 + mi * 16 + lhalf * 8 + r;
        int gcol = n0 + wn * 64 + ni * 16 + lrow;
        float v = acc[mi][ni][r] * scale;
        if (mask) v = mask[(size_t)grow * N + gcol] ? v : negFill;
        if (bias) v += bias[gcol];
        if constexpr (sizeof(OT) == 2)
          Cg[(size_t)grow * ldc + gcol] = (OT)f2bf(v);
        else
          Cg[(size_t)grow * ldc + gcol] = (OT)v;
      }
}

// ---------------------------------------------------------------------------
// fp32 -> bf16 bulk convert (2048 elements per block).
// ---------------------------------------------------------------------------
__global__ __launch_bounds__(256) void f32_to_bf16_kernel(
    const float* __restrict__ x, unsigned short* __restrict__ y)
{
  size_t i = ((size_t)blockIdx.x * 256 + threadIdx.x) * 8;
  f32x4 a = *reinterpret_cast<const f32x4*>(x + i);
  f32x4 b = *reinterpret_cast<const f32x4*>(x + i + 4);
  u16x8 o;
#pragma unroll
  for (int j = 0; j < 4; ++j) { o[j] = f2bf(a[j]); o[4 + j] = f2bf(b[j]); }
  *reinterpret_cast<u16x8*>(y + i) = o;
}

// ---------------------------------------------------------------------------
// y = x + tanh(gate) * (depthwise_conv3_seq(x) + cbias)      (bf16 in/out)
// ---------------------------------------------------------------------------
__global__ __launch_bounds__(256) void conv_gate_kernel(
    const unsigned short* __restrict__ x, unsigned short* __restrict__ y,
    const float* __restrict__ ker, const float* __restrict__ cbias,
    const float* __restrict__ gate, int S, int D)
{
  size_t idx = (size_t)blockIdx.x * 256 + threadIdx.x;
  int d = (int)(idx % D);
  int s = (int)((idx / D) % S);
  float tg = tanhf(gate[0]);
  float xm = (s > 0)     ? bf2f(x[idx - D]) : 0.0f;
  float xc = bf2f(x[idx]);
  float xp = (s < S - 1) ? bf2f(x[idx + D]) : 0.0f;
  float c = xm * ker[d * 3 + 0] + xc * ker[d * 3 + 1] + xp * ker[d * 3 + 2] + cbias[d];
  y[idx] = f2bf(xc + tg * c);
}

// ---------------------------------------------------------------------------
// Same conv, but stores the result transposed: yT[b,d,s] (tiled via LDS),
// so P@V can use the row-major [N,K] WMMA B path.
// ---------------------------------------------------------------------------
__global__ __launch_bounds__(256) void conv_gate_transpose_kernel(
    const unsigned short* __restrict__ x,   // [B,S,D]
    unsigned short* __restrict__ yT,        // [B,D,S]
    const float* __restrict__ ker, const float* __restrict__ cbias,
    const float* __restrict__ gate, int S, int D)
{
  __shared__ unsigned short tile[64][65];
  const int b = blockIdx.z;
  const int s0 = blockIdx.x * 64, d0 = blockIdx.y * 64;
  const int t = threadIdx.x;
  const float tg = tanhf(gate[0]);
  const unsigned short* xb = x + (size_t)b * S * D;
#pragma unroll
  for (int it = 0; it < 16; ++it) {
    int lin = it * 256 + t;
    int si = lin >> 6, di = lin & 63;
    int s = s0 + si, d = d0 + di;
    size_t idx = (size_t)s * D + d;
    float xm = (s > 0)     ? bf2f(xb[idx - D]) : 0.0f;
    float xc = bf2f(xb[idx]);
    float xp = (s < S - 1) ? bf2f(xb[idx + D]) : 0.0f;
    float c = xm * ker[d * 3 + 0] + xc * ker[d * 3 + 1] + xp * ker[d * 3 + 2] + cbias[d];
    tile[si][di] = f2bf(xc + tg * c);
  }
  __syncthreads();
  unsigned short* yb = yT + (size_t)b * S * D;
#pragma unroll
  for (int it = 0; it < 4; ++it) {
    int lin = (it * 256 + t) * 4;
    int di = lin >> 6, si = lin & 63;
    u16x4 v;
#pragma unroll
    for (int j = 0; j < 4; ++j) v[j] = tile[si + j][di];
    *reinterpret_cast<u16x4*>(yb + (size_t)(d0 + di) * S + s0 + si) = v;
  }
}

// ---------------------------------------------------------------------------
// Row softmax over 2048 fp32 logits; writes normalized bf16 P packed into the
// front half of the same row's storage (consumed later with lda = 2*cols).
// ---------------------------------------------------------------------------
__global__ __launch_bounds__(256) void softmax_rows_kernel(float* __restrict__ p, int cols)
{
  __shared__ float red[256];
  float* row = p + (size_t)blockIdx.x * cols;
  unsigned short* orow = (unsigned short*)p + (size_t)blockIdx.x * (cols * 2);
  const int t = threadIdx.x;

  float vmax = -3.402823466e38f;
#pragma unroll
  for (int j = 0; j < 8; ++j) vmax = fmaxf(vmax, row[t + j * 256]);
  red[t] = vmax; __syncthreads();
  for (int s = 128; s > 0; s >>= 1) { if (t < s) red[t] = fmaxf(red[t], red[t + s]); __syncthreads(); }
  float m = red[0]; __syncthreads();

  float ev[8];
  float lsum = 0.0f;
#pragma unroll
  for (int j = 0; j < 8; ++j) { ev[j] = __expf(row[t + j * 256] - m); lsum += ev[j]; }
  red[t] = lsum; __syncthreads();
  for (int s = 128; s > 0; s >>= 1) { if (t < s) red[t] += red[t + s]; __syncthreads(); }
  float inv = 1.0f / red[0];
  __syncthreads();
#pragma unroll
  for (int j = 0; j < 8; ++j) orow[t + j * 256] = f2bf(ev[j] * inv);
}

// ---------------------------------------------------------------------------
extern "C" void kernel_launch(void* const* d_in, const int* in_sizes, int n_in,
                              void* d_out, int out_size, void* d_ws, size_t ws_size,
                              hipStream_t stream)
{
  (void)in_sizes; (void)n_in; (void)out_size; (void)ws_size;
  constexpr int B = 4, SQ = 2048, SK = 2048, D = 1024;

  const float* qh   = (const float*)d_in[0];
  const float* mh   = (const float*)d_in[1];
  const unsigned char* amask = (const unsigned char*)d_in[2];
  const float* q_w  = (const float*)d_in[3];
  const float* q_b  = (const float*)d_in[4];
  const float* k_w  = (const float*)d_in[5];
  const float* k_b  = (const float*)d_in[6];
  const float* v_w  = (const float*)d_in[7];
  const float* v_b  = (const float*)d_in[8];
  const float* o_w  = (const float*)d_in[9];
  const float* o_b  = (const float*)d_in[10];
  const float* q_ker = (const float*)d_in[11];
  const float* q_cb  = (const float*)d_in[12];
  const float* q_g   = (const float*)d_in[13];
  const float* k_ker = (const float*)d_in[14];
  const float* k_cb  = (const float*)d_in[15];
  const float* k_g   = (const float*)d_in[16];
  const float* v_ker = (const float*)d_in[17];
  const float* v_cb  = (const float*)d_in[18];
  const float* v_g   = (const float*)d_in[19];

  // Workspace (MB offsets), everything L2-resident on 192MB L2:
  //  0 qhb(16) | 16 mhb(16) | 32 wq(2) wk(2) wv(2) wo(2) | 40 qc(16) | 56 kc(16)
  //  72 vT(16) | 88 logits f32 / packed bf16 P (64) | 152 raw(16) | 168 att(16)
  char* ws = (char*)d_ws;
  unsigned short* qhb = (unsigned short*)(ws);
  unsigned short* mhb = (unsigned short*)(ws + ((size_t)16 << 20));
  unsigned short* wq  = (unsigned short*)(ws + ((size_t)32 << 20));
  unsigned short* wk  = (unsigned short*)(ws + ((size_t)34 << 20));
  unsigned short* wv  = (unsigned short*)(ws + ((size_t)36 << 20));
  unsigned short* wo  = (unsigned short*)(ws + ((size_t)38 << 20));
  unsigned short* qc  = (unsigned short*)(ws + ((size_t)40 << 20));
  unsigned short* kc  = (unsigned short*)(ws + ((size_t)56 << 20));
  unsigned short* vT  = (unsigned short*)(ws + ((size_t)72 << 20));
  float*          lgt = (float*)         (ws + ((size_t)88 << 20));
  unsigned short* raw = (unsigned short*)(ws + ((size_t)152 << 20));
  unsigned short* att = (unsigned short*)(ws + ((size_t)168 << 20));

  const float NEG = -3.402823466e38f;   // finfo(float32).min
  const float ISQ = 0.03125f;           // 1/sqrt(1024)
  dim3 blk(256);

  // ---- one-time fp32 -> bf16 conversions (activations + weights) ----
  f32_to_bf16_kernel<<<(B * SQ * D) / 2048, blk, 0, stream>>>(qh, qhb);
  f32_to_bf16_kernel<<<(B * SK * D) / 2048, blk, 0, stream>>>(mh, mhb);
  f32_to_bf16_kernel<<<(D * D) / 2048, blk, 0, stream>>>(q_w, wq);
  f32_to_bf16_kernel<<<(D * D) / 2048, blk, 0, stream>>>(k_w, wk);
  f32_to_bf16_kernel<<<(D * D) / 2048, blk, 0, stream>>>(v_w, wv);
  f32_to_bf16_kernel<<<(D * D) / 2048, blk, 0, stream>>>(o_w, wo);

  dim3 gProj(D / 128, (B * SQ) / 256, 1);       // 8 x 32
  dim3 gConv((B * SQ * D) / 256);

  // Q = conv(qh @ q_w^T + q_b)
  wmma_gemm_bf16<unsigned short><<<gProj, blk, 0, stream>>>(
      qhb, 0, D, wq, 0, D, raw, 0, D, D, D, q_b, nullptr, 1.0f, NEG);
  conv_gate_kernel<<<gConv, blk, 0, stream>>>(raw, qc, q_ker, q_cb, q_g, SQ, D);

  // K = conv(mh @ k_w^T + k_b)
  wmma_gemm_bf16<unsigned short><<<gProj, blk, 0, stream>>>(
      mhb, 0, D, wk, 0, D, raw, 0, D, D, D, k_b, nullptr, 1.0f, NEG);
  conv_gate_kernel<<<gConv, blk, 0, stream>>>(raw, kc, k_ker, k_cb, k_g, SK, D);

  // V = conv(mh @ v_w^T + v_b), stored transposed [B,D,SK]
  wmma_gemm_bf16<unsigned short><<<gProj, blk, 0, stream>>>(
      mhb, 0, D, wv, 0, D, raw, 0, D, D, D, v_b, nullptr, 1.0f, NEG);
  dim3 gConvT(SK / 64, D / 64, B);
  conv_gate_transpose_kernel<<<gConvT, blk, 0, stream>>>(raw, vT, v_ker, v_cb, v_g, SK, D);

  // logits = mask ? (q @ k^T)/sqrt(D) : -FLT_MAX
  dim3 gLog(SK / 128, SQ / 256, B);             // 16 x 8 x 4
  wmma_gemm_bf16<float><<<gLog, blk, 0, stream>>>(
      qc, (long)SQ * D, D, kc, (long)SK * D, D, lgt, (long)SQ * SK, SK,
      SK, D, nullptr, amask, ISQ, NEG);

  // softmax rows -> packed bf16 P in place
  softmax_rows_kernel<<<B * SQ, blk, 0, stream>>>(lgt, SK);

  // att = P @ V   (A = packed bf16 P, lda = 2*SK; B = vT [D,SK] row-major)
  dim3 gPV(D / 128, SQ / 256, B);               // 8 x 8 x 4
  wmma_gemm_bf16<unsigned short><<<gPV, blk, 0, stream>>>(
      (const unsigned short*)lgt, (long)SQ * 2 * SK, 2 * SK,
      vT, (long)D * SK, SK,
      att, (long)SQ * D, D,
      D, SK, nullptr, nullptr, 1.0f, NEG);

  // out = att @ o_w^T + o_b   (fp32 output)
  wmma_gemm_bf16<float><<<gProj, blk, 0, stream>>>(
      att, 0, D, wo, 0, D, (float*)d_out, 0, D, D, D, o_b, nullptr, 1.0f, NEG);
}
